// LocalfeatureIntegrationTransformer3D_49512382988388
// MI455X (gfx1250) — compile-verified, matrix operand order checked by reading, not run on hardware
//
#include <hip/hip_runtime.h>

typedef unsigned short u16;
typedef __attribute__((ext_vector_type(16))) __bf16 v16bf;
typedef __attribute__((ext_vector_type(8)))  float  v8f;

union FragBF { v16bf v; uint4 q[2]; };

__device__ __forceinline__ float bf2f(u16 h) {
  unsigned u = ((unsigned)h) << 16;
  return __builtin_bit_cast(float, u);
}
__device__ __forceinline__ u16 f2bf(float f) {
  unsigned u = __builtin_bit_cast(unsigned, f);
  u += 0x7FFFu + ((u >> 16) & 1u);   // round-to-nearest-even
  return (u16)(u >> 16);
}

// ---------------------------------------------------------------------------
// Generic bf16 GEMM: C[M][N] = A[M][K] (row-major bf16) x Bt[N][K] (bf16)
// Block: 256 thr = 8 waves; tile 64(M) x 128(N); wave = 16x64 (4 n-subtiles).
// WMMA fragment layouts per CDNA5 ISA 7.12.2 (wave32, 16x16x32 bf16):
//   A: lane l holds row m=(l&15); halves j<8 -> K=kh+j, j>=8 -> K=16+kh+j-8,
//      kh=8*(l>>4).  B symmetric with N=(l&15).  C: M=r+8*(l>>4), N=(l&15).
// ---------------------------------------------------------------------------
constexpr int EPI_BF16 = 0, EPI_BF16_T = 1, EPI_BF16_SCALE = 2,
              EPI_GELU_BF16 = 3, EPI_BIAS_ACC_F32 = 4, EPI_ATOMIC_F32 = 5;

template<int EPI>
__global__ __launch_bounds__(256) void gemm_bt_kernel(
    const u16* __restrict__ A, int lda,
    const u16* __restrict__ Bt, int ldb,
    int N, int Ksplit,
    int rows_per_batch, long long bt_batch_stride,
    const float* __restrict__ bias, float scale,
    void* __restrict__ outp, int out_rb)
{
  __shared__ u16 Asm[64 * 40];
  const int m0   = blockIdx.x * 64;
  const int n0   = blockIdx.y * 128;
  const int kbeg = blockIdx.z * Ksplit;
  const int w    = threadIdx.x >> 5;
  const int lane = threadIdx.x & 31;
  const int msub = (w & 3) * 16;
  const int nsub = (w >> 2) * 64;
  const int hl   = lane >> 4;
  const int l15  = lane & 15;
  const int kh   = hl * 8;
  const int b    = m0 / rows_per_batch;
  const u16* BtB = Bt + (long long)b * bt_batch_stride;

  v8f zero = {0.f,0.f,0.f,0.f,0.f,0.f,0.f,0.f};
  v8f acc[4];
  #pragma unroll
  for (int j = 0; j < 4; ++j) acc[j] = zero;

  const int sr = threadIdx.x >> 2;
  const int sc = (threadIdx.x & 3) * 8;

  for (int kk = kbeg; kk < kbeg + Ksplit; kk += 32) {
    *(uint4*)&Asm[sr * 40 + sc] =
        *(const uint4*)(A + (long long)(m0 + sr) * lda + kk + sc);
    __syncthreads();
    FragBF a;
    const u16* ap = &Asm[(msub + l15) * 40 + kh];
    a.q[0] = *(const uint4*)ap;
    a.q[1] = *(const uint4*)(ap + 16);
    #pragma unroll
    for (int j = 0; j < 4; ++j) {
      FragBF bfr;
      const u16* bp = BtB + (long long)(n0 + nsub + j * 16 + l15) * ldb + kk + kh;
      bfr.q[0] = *(const uint4*)bp;
      bfr.q[1] = *(const uint4*)(bp + 16);
      acc[j] = __builtin_amdgcn_wmma_f32_16x16x32_bf16(
          false, a.v, false, bfr.v, (short)0, acc[j], false, false);
    }
    __syncthreads();
  }

  #pragma unroll
  for (int j = 0; j < 4; ++j) {
    #pragma unroll
    for (int r = 0; r < 8; ++r) {
      const int m = m0 + msub + r + 8 * hl;
      const int n = n0 + nsub + j * 16 + l15;
      float v = acc[j][r];
      if (EPI == EPI_BF16) {
        ((u16*)outp)[(long long)m * N + n] = f2bf(v);
      } else if (EPI == EPI_BF16_T) {
        const int bo = m / out_rb, ml = m % out_rb;
        ((u16*)outp)[((long long)bo * N + n) * out_rb + ml] = f2bf(v);
      } else if (EPI == EPI_BF16_SCALE) {
        ((u16*)outp)[(long long)m * N + n] = f2bf(v * scale);
      } else if (EPI == EPI_GELU_BF16) {
        float xg = v + bias[n];
        float g = 0.5f * xg * (1.0f + erff(xg * 0.70710678118654752f));
        ((u16*)outp)[(long long)m * N + n] = f2bf(g);
      } else if (EPI == EPI_BIAS_ACC_F32) {
        ((float*)outp)[(long long)m * N + n] += v + bias[n];
      } else { // EPI_ATOMIC_F32 (split-K)
        atomicAdd(&((float*)outp)[(long long)m * N + n], v);
      }
    }
  }
}

// ---------------------------------------------------------------------------
// Conv3d stride-2 VALID as implicit GEMM: M=32768 tokens, N=256, K=3456.
// K = c*27 + tap matches conv_w's native flat layout.  A gathered from f32 x,
// converted to bf16 into LDS; weights streamed from L2 in Bt layout.
// Epilogue: +bias, ReLU, store h f32 [token][256].
// ---------------------------------------------------------------------------
__global__ __launch_bounds__(256) void conv_gemm_kernel(
    const float* __restrict__ x, const u16* __restrict__ Wct,
    const float* __restrict__ bias, float* __restrict__ h)
{
  __shared__ u16 Asm[64 * 40];
  const int m0   = blockIdx.x * 64;
  const int w    = threadIdx.x >> 5, lane = threadIdx.x & 31;
  const int msub = (w & 3) * 16, nhalf = (w >> 2) * 128;
  const int hl   = lane >> 4, l15 = lane & 15, kh = hl * 8;

  v8f zero = {0.f,0.f,0.f,0.f,0.f,0.f,0.f,0.f};
  v8f acc[8];
  #pragma unroll
  for (int j = 0; j < 8; ++j) acc[j] = zero;

  const int e0 = threadIdx.x * 8;
  const int kl = e0 >> 6;          // k-lane within 32-chunk (uniform per thread)
  const int mbase = e0 & 63;
  int pb[8], ps0[8], ps1[8], ps2[8];
  #pragma unroll
  for (int i = 0; i < 8; ++i) {
    int tg = m0 + mbase + i;
    pb[i] = tg >> 12;
    int l = tg & 4095;
    ps0[i] = l >> 8; ps1[i] = (l >> 4) & 15; ps2[i] = l & 15;
  }

  for (int k0 = 0; k0 < 3456; k0 += 32) {
    int k = k0 + kl;
    int c = k / 27;
    int tap = k - c * 27;
    int t0 = tap / 9, rm = tap - t0 * 9;
    int t1 = rm / 3, t2 = rm - t1 * 3;
    #pragma unroll
    for (int i = 0; i < 8; ++i) {
      long long idx = ((((long long)pb[i] * 128 + c) * 33 + (2 * ps0[i] + t0)) * 33
                       + (2 * ps1[i] + t1)) * 33 + (2 * ps2[i] + t2);
      Asm[(mbase + i) * 40 + kl] = f2bf(x[idx]);
    }
    __syncthreads();
    FragBF a;
    const u16* ap = &Asm[(msub + l15) * 40 + kh];
    a.q[0] = *(const uint4*)ap;
    a.q[1] = *(const uint4*)(ap + 16);
    #pragma unroll
    for (int j = 0; j < 8; ++j) {
      FragBF bfr;
      const u16* bp = Wct + (long long)(nhalf + j * 16 + l15) * 3456 + k0 + kh;
      bfr.q[0] = *(const uint4*)bp;
      bfr.q[1] = *(const uint4*)(bp + 16);
      acc[j] = __builtin_amdgcn_wmma_f32_16x16x32_bf16(
          false, a.v, false, bfr.v, (short)0, acc[j], false, false);
    }
    __syncthreads();
  }

  #pragma unroll
  for (int j = 0; j < 8; ++j) {
    #pragma unroll
    for (int r = 0; r < 8; ++r) {
      int m = m0 + msub + r + 8 * hl;
      int o = nhalf + j * 16 + l15;
      float v = acc[j][r] + bias[o];
      h[(long long)m * 256 + o] = fmaxf(v, 0.f);
    }
  }
}

// ---------------------------------------------------------------------------
// Logits + softmax(N=64) + column-sum: per batch, per 128-token tile.
// C[l][n] = k_bf[l][:] . q_bf[n][:]; softmax over n via 16-lane shfl_xor;
// +1e-8; write attn^T bf16 [b][n][l]; atomic colsum[b][n] for renorm-over-L.
// ---------------------------------------------------------------------------
__global__ __launch_bounds__(256) void logits_softmax_kernel(
    const u16* __restrict__ kmat, const u16* __restrict__ q,
    u16* __restrict__ attn_t, float* __restrict__ colsum)
{
  __shared__ u16 Asm[128 * 40];
  __shared__ float cs[64];
  const int b  = blockIdx.y;
  const int m0 = blockIdx.x * 128;
  const int w  = threadIdx.x >> 5, lane = threadIdx.x & 31;
  const int hl = lane >> 4, l15 = lane & 15, kh = hl * 8;
  const u16* A = kmat + (long long)b * 4096 * 256;
  const u16* Q = q    + (long long)b * 64 * 256;

  v8f zero = {0.f,0.f,0.f,0.f,0.f,0.f,0.f,0.f};
  v8f acc[4];
  #pragma unroll
  for (int j = 0; j < 4; ++j) acc[j] = zero;

  const int sr = threadIdx.x >> 2;
  const int sc = (threadIdx.x & 3) * 8;

  for (int k0 = 0; k0 < 256; k0 += 32) {
    #pragma unroll
    for (int rr = sr; rr < 128; rr += 64)
      *(uint4*)&Asm[rr * 40 + sc] =
          *(const uint4*)(A + (long long)(m0 + rr) * 256 + k0 + sc);
    __syncthreads();
    FragBF a;
    const u16* ap = &Asm[(w * 16 + l15) * 40 + kh];
    a.q[0] = *(const uint4*)ap;
    a.q[1] = *(const uint4*)(ap + 16);
    #pragma unroll
    for (int j = 0; j < 4; ++j) {
      FragBF bfr;
      const u16* bp = Q + (long long)(j * 16 + l15) * 256 + k0 + kh;
      bfr.q[0] = *(const uint4*)bp;
      bfr.q[1] = *(const uint4*)(bp + 16);
      acc[j] = __builtin_amdgcn_wmma_f32_16x16x32_bf16(
          false, a.v, false, bfr.v, (short)0, acc[j], false, false);
    }
    __syncthreads();
  }

  if (threadIdx.x < 64) cs[threadIdx.x] = 0.f;
  __syncthreads();

  float csl[4] = {0.f, 0.f, 0.f, 0.f};
  #pragma unroll
  for (int r = 0; r < 8; ++r) {
    float v0 = acc[0][r], v1 = acc[1][r], v2 = acc[2][r], v3 = acc[3][r];
    float mx = fmaxf(fmaxf(v0, v1), fmaxf(v2, v3));
    for (int off = 1; off < 16; off <<= 1) mx = fmaxf(mx, __shfl_xor(mx, off, 32));
    v0 = __expf(v0 - mx); v1 = __expf(v1 - mx);
    v2 = __expf(v2 - mx); v3 = __expf(v3 - mx);
    float sm = v0 + v1 + v2 + v3;
    for (int off = 1; off < 16; off <<= 1) sm += __shfl_xor(sm, off, 32);
    float rinv = 1.f / sm;
    v0 = v0 * rinv + 1e-8f; v1 = v1 * rinv + 1e-8f;
    v2 = v2 * rinv + 1e-8f; v3 = v3 * rinv + 1e-8f;
    const int l = m0 + w * 16 + r + 8 * hl;
    attn_t[((long long)b * 64 + ( 0 + l15)) * 4096 + l] = f2bf(v0);
    attn_t[((long long)b * 64 + (16 + l15)) * 4096 + l] = f2bf(v1);
    attn_t[((long long)b * 64 + (32 + l15)) * 4096 + l] = f2bf(v2);
    attn_t[((long long)b * 64 + (48 + l15)) * 4096 + l] = f2bf(v3);
    csl[0] += v0; csl[1] += v1; csl[2] += v2; csl[3] += v3;
  }
  #pragma unroll
  for (int j = 0; j < 4; ++j) atomicAdd(&cs[j * 16 + l15], csl[j]);
  __syncthreads();
  if (threadIdx.x < 64) atomicAdd(&colsum[b * 64 + threadIdx.x], cs[threadIdx.x]);
}

// ---------------------------------------------------------------------------
// LayerNorm over 256 channels, one wave per row, f32 in -> bf16 out.
// ---------------------------------------------------------------------------
__global__ __launch_bounds__(256) void ln_rows_kernel(
    const float* __restrict__ in, const float* __restrict__ g,
    const float* __restrict__ bb, u16* __restrict__ out, int R)
{
  const int w = threadIdx.x >> 5, lane = threadIdx.x & 31;
  const int row = blockIdx.x * 8 + w;
  if (row >= R) return;
  const float* p = in + (long long)row * 256;
  float vals[8], s = 0.f, s2 = 0.f;
  #pragma unroll
  for (int i = 0; i < 8; ++i) {
    vals[i] = p[lane + i * 32];
    s += vals[i]; s2 += vals[i] * vals[i];
  }
  for (int off = 16; off; off >>= 1) {
    s  += __shfl_xor(s,  off, 32);
    s2 += __shfl_xor(s2, off, 32);
  }
  float mu  = s * (1.f / 256.f);
  float var = s2 * (1.f / 256.f) - mu * mu;
  float inv = rsqrtf(var + 1e-5f);
  #pragma unroll
  for (int i = 0; i < 8; ++i) {
    int c = lane + i * 32;
    out[(long long)row * 256 + c] = f2bf((vals[i] - mu) * inv * g[c] + bb[c]);
  }
}

// ------------------------------ small utilities ----------------------------
__global__ void cast_bf_kernel(const float* __restrict__ in, u16* __restrict__ out, int n) {
  int i = blockIdx.x * 256 + threadIdx.x;
  if (i < n) out[i] = f2bf(in[i]);
}
__global__ void transpose_bf_kernel(const float* __restrict__ in, u16* __restrict__ out,
                                    int K, int N) {   // in[K][N] -> out[N][K]
  int i = blockIdx.x * 256 + threadIdx.x;
  if (i < K * N) {
    int nn = i / K, k = i - nn * K;
    out[i] = f2bf(in[(long long)k * N + nn]);
  }
}
__global__ void zero_f32_kernel(float* p, int n) {
  int i = blockIdx.x * 256 + threadIdx.x;
  if (i < n) p[i] = 0.f;
}
__global__ void bcast_kernel(const float* __restrict__ in, float* __restrict__ out,
                             int n, int period) {
  int i = blockIdx.x * 256 + threadIdx.x;
  if (i < n) out[i] = in[i % period];
}
__global__ void apply_update_kernel(float* __restrict__ tmpl, const float* __restrict__ cbuf,
                                    const float* __restrict__ colsum, int n) {
  int i = blockIdx.x * 256 + threadIdx.x;
  if (i < n) tmpl[i] += cbuf[i] / colsum[i >> 8];
}
__global__ void final_out_kernel(const float* __restrict__ tmpl, const u16* __restrict__ attn,
                                 const float* __restrict__ colsum, float* __restrict__ out) {
  int i = blockIdx.x * 256 + threadIdx.x;
  if (i < 131072) {                       // templates_out (B, dim, N)
    int b = i >> 14, rem = i & 16383, d = rem >> 6, n = rem & 63;
    out[i] = tmpl[((b * 64 + n) << 8) + d];
  } else if (i < 131072 + 2097152) {      // attn_out (B, N, 16,16,16)
    int j = i - 131072;
    out[i] = bf2f(attn[j]) / colsum[j >> 12];
  }
}

// ---------------------------------------------------------------------------
extern "C" void kernel_launch(void* const* d_in, const int* in_sizes, int n_in,
                              void* d_out, int out_size, void* d_ws, size_t ws_size,
                              hipStream_t stream) {
  const float* x       = (const float*)d_in[0];
  const float* tinit   = (const float*)d_in[1];
  const float* conv_w  = (const float*)d_in[2];
  const float* conv_b  = (const float*)d_in[3];
  const float* Wq      = (const float*)d_in[4];
  const float* Wk      = (const float*)d_in[5];
  const float* Wv      = (const float*)d_in[6];
  const float* ln_in_g = (const float*)d_in[7];
  const float* ln_in_b = (const float*)d_in[8];
  const float* ln_t_g  = (const float*)d_in[9];
  const float* ln_t_b  = (const float*)d_in[10];
  const float* ln_m_g  = (const float*)d_in[11];
  const float* ln_m_b  = (const float*)d_in[12];
  const float* W1      = (const float*)d_in[13];
  const float* b1      = (const float*)d_in[14];
  const float* W2      = (const float*)d_in[15];
  const float* b2      = (const float*)d_in[16];
  float* out = (float*)d_out;

  char* ws = (char*)d_ws;
  size_t off = 0;
  auto alloc = [&](size_t bytes) -> void* {
    off = (off + 255) & ~(size_t)255;
    void* p = ws + off;
    off += bytes;
    return p;
  };
  u16*   Wct  = (u16*)  alloc(884736ull * 2);   // [256][3456]
  u16*   Wqt  = (u16*)  alloc(65536ull * 2);    // [256][256]
  u16*   Wkt  = (u16*)  alloc(65536ull * 2);
  u16*   Wvt  = (u16*)  alloc(65536ull * 2);
  u16*   W1t  = (u16*)  alloc(131072ull * 2);   // [512][256]
  u16*   W2t  = (u16*)  alloc(131072ull * 2);   // [256][512]
  float* hbuf = (float*)alloc(8388608ull * 4);  // conv out (32768,256) f32
  u16*   tok  = (u16*)  alloc(8388608ull * 2);  // LN(tokens) bf16
  u16*   kbf  = (u16*)  alloc(8388608ull * 2);  // k (B,4096,256)
  u16*   vtb  = (u16*)  alloc(8388608ull * 2);  // v^T (B,256,4096)
  float* tmpl = (float*)alloc(131072ull * 4);   // templates (B,64,256)
  u16*   tbf  = (u16*)  alloc(131072ull * 2);   // LN(templates) bf16 (t and m)
  u16*   qbf  = (u16*)  alloc(131072ull * 2);   // q (B,64,256)
  u16*   attn = (u16*)  alloc(2097152ull * 2);  // attn^T (B,64,4096)
  float* csum = (float*)alloc(512ull * 4);      // colsum (B,64)
  u16*   h1b  = (u16*)  alloc(262144ull * 2);   // MLP hidden (512,512)
  float* cbuf = (float*)alloc(131072ull * 4);   // split-K accum (512,256)

  const int HUGE_RB = 1 << 30;

  // weight conversion (conv_w flat layout already matches Bt[K-ordering])
  cast_bf_kernel<<<3456, 256, 0, stream>>>(conv_w, Wct, 884736);
  transpose_bf_kernel<<<256, 256, 0, stream>>>(Wq, Wqt, 256, 256);
  transpose_bf_kernel<<<256, 256, 0, stream>>>(Wk, Wkt, 256, 256);
  transpose_bf_kernel<<<256, 256, 0, stream>>>(Wv, Wvt, 256, 256);
  transpose_bf_kernel<<<512, 256, 0, stream>>>(W1, W1t, 256, 512);
  transpose_bf_kernel<<<512, 256, 0, stream>>>(W2, W2t, 512, 256);

  // conv + relu (implicit GEMM, 58 GFLOP)
  conv_gemm_kernel<<<512, 256, 0, stream>>>(x, Wct, conv_b, hbuf);
  // token layernorm -> bf16
  ln_rows_kernel<<<4096, 256, 0, stream>>>(hbuf, ln_in_g, ln_in_b, tok, 32768);

  // k = tok @ Wk ; v^T = (tok @ Wv)^T
  gemm_bt_kernel<EPI_BF16><<<dim3(512, 2, 1), 256, 0, stream>>>(
      tok, 256, Wkt, 256, 256, 256, HUGE_RB, 0, nullptr, 1.f, kbf, 1);
  gemm_bt_kernel<EPI_BF16_T><<<dim3(512, 2, 1), 256, 0, stream>>>(
      tok, 256, Wvt, 256, 256, 256, HUGE_RB, 0, nullptr, 1.f, vtb, 4096);

  // templates = broadcast(templates_init)
  bcast_kernel<<<512, 256, 0, stream>>>(tinit, tmpl, 131072, 16384);

  for (int it = 0; it < 6; ++it) {
    // t = LN(templates);  q = (t @ Wq) * dim^-0.5
    ln_rows_kernel<<<64, 256, 0, stream>>>(tmpl, ln_t_g, ln_t_b, tbf, 512);
    gemm_bt_kernel<EPI_BF16_SCALE><<<dim3(8, 2, 1), 256, 0, stream>>>(
        tbf, 256, Wqt, 256, 256, 256, HUGE_RB, 0, nullptr, 0.0625f, qbf, 1);

    // logits + softmax(N) + 1e-8 + colsum over L
    zero_f32_kernel<<<2, 256, 0, stream>>>(csum, 512);
    logits_softmax_kernel<<<dim3(32, 8, 1), 256, 0, stream>>>(kbf, qbf, attn, csum);

    // update = attn^T @ v  (split-K x8, f32 atomics), then templates += upd/colsum
    zero_f32_kernel<<<512, 256, 0, stream>>>(cbuf, 131072);
    gemm_bt_kernel<EPI_ATOMIC_F32><<<dim3(8, 2, 8), 256, 0, stream>>>(
        attn, 4096, vtb, 4096, 256, 512, 64, 1048576ll, nullptr, 1.f, cbuf, 1);
    apply_update_kernel<<<512, 256, 0, stream>>>(tmpl, cbuf, csum, 131072);

    // MLP: m = LN(templates); templates += GELU(m@W1+b1)@W2 + b2
    ln_rows_kernel<<<64, 256, 0, stream>>>(tmpl, ln_m_g, ln_m_b, tbf, 512);
    gemm_bt_kernel<EPI_GELU_BF16><<<dim3(8, 4, 1), 256, 0, stream>>>(
        tbf, 256, W1t, 256, 512, 256, HUGE_RB, 0, b1, 1.f, h1b, 1);
    gemm_bt_kernel<EPI_BIAS_ACC_F32><<<dim3(8, 2, 1), 256, 0, stream>>>(
        h1b, 512, W2t, 512, 256, 512, HUGE_RB, 0, b2, 1.f, tmpl, 1);
  }

  // outputs: templates^T (B,dim,N) then attn/colsum as (B,N,16,16,16)
  final_out_kernel<<<8704, 256, 0, stream>>>(tmpl, attn, csum, out);
}
